// IntraEncoderDecoder_31997506355543
// MI455X (gfx1250) — compile-verified, matrix operand order checked by reading, not run on hardware
//
#include <hip/hip_runtime.h>

// ---------------------------------------------------------------------------
// MI455X (gfx1250) implementation of the bi-LSTM encoder + intra-attention
// decoder + vocab projection.
//
// Roofline: the vocab projection [1024,3072]x[3072,32000] is 200 of ~275 total
// GFLOP -> hoisted out of the sequential scan as one bf16 WMMA GEMM. GEMM block
// tile is 64(M) x 128(N): each W_v B-fragment feeds 4 WMMAs, so worst-case W
// traffic is 3.1GB and, with co-resident M-blocks sharing lines in the 192MB
// L2, effective HBM traffic approaches the single 196MB pass (~8.4us @ 23.3
// TB/s) -> compute-bound on v_wmma_f32_16x16x32_bf16. Sequential h@Whh
// recurrences run as persistent single-workgroup kernels (32 waves) with h in
// LDS (bf16) and cell state c in registers in the native WMMA C layout.
// ---------------------------------------------------------------------------

#define DEV __device__ __forceinline__

typedef __attribute__((ext_vector_type(16))) __bf16 v16bf;
typedef __attribute__((ext_vector_type(8)))  __bf16 v8bf;
typedef __attribute__((ext_vector_type(8)))  float  v8f;

constexpr int CB  = 16;     // batch
constexpr int CTE = 400;    // encoder steps
constexpr int CTD = 64;     // decoder steps
constexpr int CE  = 256;    // emb dim
constexpr int CH  = 512;    // encoder hidden
constexpr int CH2 = 1024;   // decoder hidden
constexpr int CV  = 32000;  // vocab
constexpr int G4H  = 4 * CH;       // 2048 encoder gates
constexpr int G4H2 = 4 * CH2;      // 4096 decoder gates
constexpr int KIN  = CE + 2 * CH2; // 2304 decoder input width
constexpr int F3   = 3 * CH2;      // 3072 feature width

DEV __bf16 f2bf(float f) {
  unsigned u = __builtin_bit_cast(unsigned, f);
  unsigned r = (u + 0x7FFFu + ((u >> 16) & 1u)) >> 16;   // RNE
  unsigned short s = (unsigned short)r;
  return __builtin_bit_cast(__bf16, s);
}
DEV float bf2f(__bf16 b) {
  unsigned short s = __builtin_bit_cast(unsigned short, b);
  unsigned u = ((unsigned)s) << 16;
  return __builtin_bit_cast(float, u);
}
DEV float sigm(float x) { return 1.f / (1.f + __expf(-x)); }

// A/B fragment per CDNA5 ISA (CDNA5 ISA, 16-bit 16x32 A layout; B mirrored):
// lane L: row = L&15, K-half = (L>>4)*8; v0..3 hold K=half+0..7, v4..7 K=half+16..23.
// Both loads are 16B contiguous in K (weights stored row-major [N,K] = W^T math).
DEV v16bf load_frag(const __bf16* base, int stride, int lane, int koff) {
  int row = lane & 15;
  int kh  = ((lane >> 4) << 3) + koff;
  const __bf16* p = base + row * stride + kh;
  v8bf lo = *(const v8bf*)(p);
  v8bf hi = *(const v8bf*)(p + 16);
  return __builtin_shufflevector(lo, hi, 0,1,2,3,4,5,6,7,8,9,10,11,12,13,14,15);
}

DEV v8f wmma_bf(v16bf a, v16bf b, v8f c) {
  return __builtin_amdgcn_wmma_f32_16x16x32_bf16(false, a, false, b, (short)0, c, false, false);
}

// ---------------------------------------------------------------------------
// elementwise converters
// ---------------------------------------------------------------------------
__global__ void cvt_bf16_kernel(const float* __restrict__ in, __bf16* __restrict__ out, long n) {
  for (long i = (long)blockIdx.x * blockDim.x + threadIdx.x; i < n; i += (long)gridDim.x * blockDim.x)
    out[i] = f2bf(in[i]);
}
// out[c*R + r] = in[r*C + c]  (transpose so GEMMs see W^T with contiguous K rows)
__global__ void cvt_bf16_T_kernel(const float* __restrict__ in, __bf16* __restrict__ out, int R, int C) {
  for (long i = (long)blockIdx.x * blockDim.x + threadIdx.x; i < (long)R * C; i += (long)gridDim.x * blockDim.x) {
    int r = (int)(i / C), c = (int)(i % C);
    out[(long)c * R + r] = f2bf(in[i]);
  }
}

// embedding gathers -> time-major bf16 [T*B, E]
__global__ void gather_kernel(const int* __restrict__ ids /*[B,T]*/, const float* __restrict__ emb,
                              __bf16* __restrict__ xs, int T) {
  long n = (long)T * CB * CE;
  for (long i = (long)blockIdx.x * blockDim.x + threadIdx.x; i < n; i += (long)gridDim.x * blockDim.x) {
    int e = (int)(i % CE);
    long row = i / CE;
    int b = (int)(row % CB);
    int t = (int)(row / CB);
    int tok = ids[b * T + t];
    xs[row * CE + e] = f2bf(emb[(long)tok * CE + e]);
  }
}

// ---------------------------------------------------------------------------
// Generic WMMA GEMM: C[M,N] = A[M,K](bf16,row-major) * W[N,K]^T (bf16) + bias
// Block = 8 waves; block tile 64(M) x 128(N); A panel (64x256) staged in LDS.
// Each wave holds 4 M-subtile accumulators so every B fragment feeds 4 WMMAs
// (4x arithmetic intensity on the weight stream). Requires M % 64 == 0.
// ---------------------------------------------------------------------------
__global__ __launch_bounds__(256) void gemm_bf16_kernel(
    const __bf16* __restrict__ A, int lda,
    const __bf16* __restrict__ W, int ldb,
    const float* __restrict__ bias,
    float* __restrict__ C, int ldc, int K) {
  __shared__ __bf16 As[64 * 256];  // 32KB
  const int lane = threadIdx.x & 31;
  const int wave = threadIdx.x >> 5;
  const int m0 = blockIdx.x * 64;
  const int n0 = (blockIdx.y * 8 + wave) * 16;
  v8f acc[4] = {};
  for (int kk = 0; kk < K; kk += 256) {
    // cooperative A panel load: 64 rows x 256 cols, 4 x 16 bf16 per thread
    for (int q = 0; q < 4; ++q) {
      int l = (threadIdx.x + q * 256) * 16;
      int r = l >> 8;
      int c = l & 255;
      const __bf16* src = A + (long)(m0 + r) * lda + kk + c;
      *(v8bf*)(&As[r * 256 + c])     = *(const v8bf*)(src);
      *(v8bf*)(&As[r * 256 + c + 8]) = *(const v8bf*)(src + 8);
    }
    __syncthreads();
    const __bf16* wrow = W + (long)n0 * ldb;
    if (kk + 256 < K) __builtin_prefetch(wrow + (lane & 15) * ldb + kk + 256, 0, 0);
    for (int ks = 0; ks < 256; ks += 32) {
      v16bf b = load_frag(wrow, ldb, lane, kk + ks);
      for (int mt = 0; mt < 4; ++mt) {
        v16bf a = load_frag(&As[mt * 16 * 256], 256, lane, ks);
        acc[mt] = wmma_bf(a, b, acc[mt]);
      }
    }
    __syncthreads();
  }
  int n = lane & 15;
  float bv = bias ? bias[n0 + n] : 0.f;
  for (int mt = 0; mt < 4; ++mt) {
    for (int j = 0; j < 8; ++j) {
      int m = mt * 16 + j + ((lane >> 4) << 3);
      C[(long)(m0 + m) * ldc + n0 + n] = acc[mt][j] + bv;
    }
  }
}

// ---------------------------------------------------------------------------
// Encoder recurrence: one workgroup per direction, 32 waves. z = xW[t] + h@Whh^T.
// Output columns (2048) = 128 tiles; wave w owns tile w of each gate (stride 32
// tiles = one gate), so i/f/g/o for the same cell index land in one wave and the
// cell state c stays in registers in WMMA C layout.
// ---------------------------------------------------------------------------
__global__ __launch_bounds__(1024) void encoder_lstm_kernel(
    const float* __restrict__ xWf, const float* __restrict__ xWb,
    const __bf16* __restrict__ Whhf, const __bf16* __restrict__ Whhb,
    const float* __restrict__ h0, const float* __restrict__ c0,
    __bf16* __restrict__ encout_bf /*[B,TE,2H] bf16*/,
    float* __restrict__ hcat, float* __restrict__ ccat /*[B,2H]*/) {
  const int dir = blockIdx.x;
  const float* xW = dir ? xWb : xWf;
  const __bf16* Whh = dir ? Whhb : Whhf;
  __shared__ __bf16 hs[16 * CH];  // 16KB
  const int lane = threadIdx.x & 31;
  const int wave = threadIdx.x >> 5;
  const int n = lane & 15;
  const int hi2 = lane >> 4;
  const int colbase = wave * 16;  // within one gate [0,512)

  for (int idx = threadIdx.x; idx < 16 * CH; idx += 1024) {
    int m = idx >> 9, c = idx & (CH - 1);
    hs[idx] = f2bf(h0[(dir * CB + m) * CH + c]);
  }
  v8f creg;
  for (int j = 0; j < 8; ++j)
    creg[j] = c0[(dir * CB + j + (hi2 << 3)) * CH + colbase + n];
  __syncthreads();

  v8f hreg = {};
  for (int t = 0; t < CTE; ++t) {
    int srcT = dir ? (CTE - 1 - t) : t;
    v8f acc[4];
    for (int g = 0; g < 4; ++g) {
      int col = g * CH + colbase + n;
      for (int j = 0; j < 8; ++j)
        acc[g][j] = xW[((long)srcT * CB + j + (hi2 << 3)) * G4H + col];  // bias folded in
    }
    for (int ks = 0; ks < CH; ks += 32) {
      v16bf a = load_frag(hs, CH, lane, ks);
      for (int g = 0; g < 4; ++g) {
        v16bf b = load_frag(Whh + (long)(g * CH + colbase) * CH, CH, lane, ks);
        acc[g] = wmma_bf(a, b, acc[g]);
      }
    }
    for (int j = 0; j < 8; ++j) {
      float i_ = sigm(acc[0][j]);
      float f_ = sigm(acc[1][j]);
      float g_ = tanhf(acc[2][j]);
      float o_ = sigm(acc[3][j]);
      float c2 = f_ * creg[j] + i_ * g_;
      creg[j] = c2;
      hreg[j] = o_ * tanhf(c2);
    }
    __syncthreads();  // all A-frag reads of hs done
    int pos = dir ? (CTE - 1 - t) : t;
    for (int j = 0; j < 8; ++j) {
      int m = j + (hi2 << 3);
      __bf16 hb = f2bf(hreg[j]);
      hs[m * CH + colbase + n] = hb;
      encout_bf[((long)m * CTE + pos) * (2 * CH) + dir * CH + colbase + n] = hb;
    }
    __syncthreads();
  }
  for (int j = 0; j < 8; ++j) {
    int m = j + (hi2 << 3);
    hcat[m * (2 * CH) + dir * CH + colbase + n] = hreg[j];
    ccat[m * (2 * CH) + dir * CH + colbase + n] = creg[j];
  }
}

// ---------------------------------------------------------------------------
// Decoder: single workgroup, 32 waves, 64 sequential steps.
// Per step: WMMA h@W_ea / h@W_da, VALU attention phases, then the gate GEMM
// (K = 2048 ctx from feat + 1024 hidden from LDS) with c in registers.
// feat[TD*B, 3072] = [h2 | ctx_e | ctx_d] (bf16) doubles as the decoder-hidden
// history for intra-decoder attention and as A for the final vocab GEMM.
// ---------------------------------------------------------------------------
__global__ __launch_bounds__(1024) void decoder_kernel(
    const float* __restrict__ embW /*[TD*B,4096] bias folded*/,
    const __bf16* __restrict__ Wihd /*[4096,2304]*/,
    const __bf16* __restrict__ Whhd /*[4096,1024]*/,
    const __bf16* __restrict__ WeaT, const __bf16* __restrict__ WdaT /*[1024,1024]*/,
    const __bf16* __restrict__ encout_bf, const float* __restrict__ hcat,
    const float* __restrict__ ccat,
    __bf16* __restrict__ feat, float* __restrict__ hWe, float* __restrict__ hWd,
    float* __restrict__ sc, float* __restrict__ accum, float* __restrict__ dsc,
    float* __restrict__ out_h, float* __restrict__ out_c) {
  __shared__ __bf16 hs[16 * CH2];  // 32KB
  __shared__ float red[CB];
  __shared__ float red2[CB];
  const int lane = threadIdx.x & 31;
  const int wave = threadIdx.x >> 5;
  const int n = lane & 15;
  const int hi2 = lane >> 4;

  for (int idx = threadIdx.x; idx < 16 * CH2; idx += 1024) hs[idx] = f2bf(hcat[idx]);
  for (int idx = threadIdx.x; idx < CB * CTE; idx += 1024) accum[idx] = 0.f;
  float creg[2][8], hv[2][8];
  for (int cg = 0; cg < 2; ++cg) {
    int col = cg * 512 + wave * 16 + n;
    for (int j = 0; j < 8; ++j) creg[cg][j] = ccat[(j + (hi2 << 3)) * CH2 + col];
  }
  __syncthreads();

  for (int i = 0; i < CTD; ++i) {
    // ---- Phase A: hWe = h@W_ea, hWd = h@W_da (128 tiles / 32 waves) ----
    for (int which = 0; which < 2; ++which) {
      const __bf16* WT = which ? WdaT : WeaT;
      float* out = which ? hWd : hWe;
      for (int tt = 0; tt < 2; ++tt) {
        int n0 = (wave + tt * 32) * 16;
        v8f acc = {};
        for (int ks = 0; ks < CH2; ks += 32) {
          v16bf a = load_frag(hs, CH2, lane, ks);
          v16bf b = load_frag(WT + (long)n0 * CH2, CH2, lane, ks);
          acc = wmma_bf(a, b, acc);
        }
        for (int j = 0; j < 8; ++j) out[(j + (hi2 << 3)) * CH2 + n0 + n] = acc[j];
      }
    }
    if (threadIdx.x < CB) { red[threadIdx.x] = 0.f; red2[threadIdx.x] = 0.f; }
    __threadfence();
    __syncthreads();
    // ---- Phase B1: intra-temporal scores, accum, per-batch sum ----
    for (int p = threadIdx.x; p < CB * CTE; p += 1024) {
      int b = p / CTE, t = p % CTE;
      const float* hw = &hWe[b * CH2];
      const __bf16* eo = &encout_bf[((long)b * CTE + t) * CH2];
      float s = 0.f;
      for (int d = 0; d < CH2; ++d) s += hw[d] * bf2f(eo[d]);
      float ex = __expf(s);
      float denom = (i == 0) ? 1.f : accum[p];
      float ep = ex / denom;
      sc[p] = ep;
      accum[p] += ex;
      atomicAdd(&red[b], ep);
    }
    __threadfence();
    __syncthreads();
    // ---- Phase B2: ctx_e -> feat[:,1024:2048]; decoder scores (t<i) ----
    for (int p = threadIdx.x; p < CB * CH2; p += 1024) {
      int b = p >> 10, d = p & (CH2 - 1);
      float inv = 1.f / red[b];
      float s = 0.f;
      for (int t = 0; t < CTE; ++t)
        s += sc[b * CTE + t] * bf2f(encout_bf[((long)b * CTE + t) * CH2 + d]);
      feat[((long)(i * CB + b)) * F3 + CH2 + d] = f2bf(s * inv);
    }
    if (i > 0) {
      for (int p = threadIdx.x; p < CB * CTD; p += 1024) {
        int b = p / CTD, t = p % CTD;
        float e = 0.f;
        if (t < i) {
          const float* hw = &hWd[b * CH2];
          const __bf16* bh = &feat[((long)(t * CB + b)) * F3];  // stored h2 history
          float s = 0.f;
          for (int d = 0; d < CH2; ++d) s += hw[d] * bf2f(bh[d]);
          e = __expf(s);
        }
        dsc[p] = e;
        atomicAdd(&red2[b], e);
      }
    }
    __threadfence();
    __syncthreads();
    // ---- Phase B3: ctx_d -> feat[:,2048:3072] ----
    for (int p = threadIdx.x; p < CB * CH2; p += 1024) {
      int b = p >> 10, d = p & (CH2 - 1);
      float s = 0.f;
      if (i > 0) {
        float inv = 1.f / red2[b];
        for (int t = 0; t < i; ++t)
          s += dsc[b * CTD + t] * bf2f(feat[((long)(t * CB + b)) * F3 + d]);
        s *= inv;
      }
      feat[((long)(i * CB + b)) * F3 + 2 * CH2 + d] = f2bf(s);
    }
    __threadfence();
    __syncthreads();
    // ---- Phase C: gate GEMM z[16,4096] and LSTM cell update ----
    const __bf16* arow = &feat[(long)i * CB * F3];  // A for ctx parts
    for (int cg = 0; cg < 2; ++cg) {
      int cidx = cg * 512 + wave * 16;  // within gate [0,1024)
      v8f acc[4];
      for (int g = 0; g < 4; ++g) {
        int col = g * CH2 + cidx + n;
        for (int j = 0; j < 8; ++j)
          acc[g][j] = embW[((long)(i * CB + j + (hi2 << 3))) * G4H2 + col];  // emb part + b_d
      }
      for (int ks = 0; ks < 2 * CH2; ks += 32) {  // ctx_e|ctx_d vs Wihd[:,256:2304]
        v16bf a = load_frag(arow + CH2, F3, lane, ks);
        for (int g = 0; g < 4; ++g) {
          v16bf b = load_frag(Wihd + (long)(g * CH2 + cidx) * KIN + CE, KIN, lane, ks);
          acc[g] = wmma_bf(a, b, acc[g]);
        }
      }
      for (int ks = 0; ks < CH2; ks += 32) {      // h vs Whhd
        v16bf a = load_frag(hs, CH2, lane, ks);
        for (int g = 0; g < 4; ++g) {
          v16bf b = load_frag(Whhd + (long)(g * CH2 + cidx) * CH2, CH2, lane, ks);
          acc[g] = wmma_bf(a, b, acc[g]);
        }
      }
      for (int j = 0; j < 8; ++j) {
        float i_ = sigm(acc[0][j]);
        float f_ = sigm(acc[1][j]);
        float g_ = tanhf(acc[2][j]);
        float o_ = sigm(acc[3][j]);
        float c2 = f_ * creg[cg][j] + i_ * g_;
        creg[cg][j] = c2;
        hv[cg][j] = o_ * tanhf(c2);
      }
    }
    __syncthreads();  // hs reads complete
    for (int cg = 0; cg < 2; ++cg) {
      int col = cg * 512 + wave * 16 + n;
      for (int j = 0; j < 8; ++j) {
        int m = j + (hi2 << 3);
        __bf16 hb = f2bf(hv[cg][j]);
        hs[m * CH2 + col] = hb;
        feat[((long)(i * CB + m)) * F3 + col] = hb;
      }
    }
    __threadfence();
    __syncthreads();
  }
  for (int cg = 0; cg < 2; ++cg) {
    int col = cg * 512 + wave * 16 + n;
    for (int j = 0; j < 8; ++j) {
      int m = j + (hi2 << 3);
      out_h[m * CH2 + col] = hv[cg][j];
      out_c[m * CH2 + col] = creg[cg][j];
    }
  }
}

// ---------------------------------------------------------------------------
// log_softmax over V per (t,b) row; writes d_out in [B,TD,V] order.
// ---------------------------------------------------------------------------
__global__ __launch_bounds__(256) void logsoftmax_kernel(const float* __restrict__ logits,
                                                         float* __restrict__ out) {
  int r = blockIdx.x;            // r = t*16 + b
  int b = r & 15, t = r >> 4;
  const float* x = logits + (long)r * CV;
  __shared__ float sred[256];
  float mx = -1e30f;
  for (int v = threadIdx.x; v < CV; v += 256) mx = fmaxf(mx, x[v]);
  sred[threadIdx.x] = mx;
  __syncthreads();
  for (int s = 128; s > 0; s >>= 1) {
    if (threadIdx.x < s) sred[threadIdx.x] = fmaxf(sred[threadIdx.x], sred[threadIdx.x + s]);
    __syncthreads();
  }
  mx = sred[0];
  __syncthreads();
  float sum = 0.f;
  for (int v = threadIdx.x; v < CV; v += 256) sum += __expf(x[v] - mx);
  sred[threadIdx.x] = sum;
  __syncthreads();
  for (int s = 128; s > 0; s >>= 1) {
    if (threadIdx.x < s) sred[threadIdx.x] += sred[threadIdx.x + s];
    __syncthreads();
  }
  float lse = mx + __logf(sred[0]);
  float* o = out + ((long)b * CTD + t) * CV;
  for (int v = threadIdx.x; v < CV; v += 256) o[v] = x[v] - lse;
}

// ---------------------------------------------------------------------------
extern "C" void kernel_launch(void* const* d_in, const int* in_sizes, int n_in,
                              void* d_out, int out_size, void* d_ws, size_t ws_size,
                              hipStream_t stream) {
  const int*   enc_ids = (const int*)d_in[0];
  const int*   dec_ids = (const int*)d_in[1];
  const float* h0    = (const float*)d_in[4];
  const float* c0    = (const float*)d_in[5];
  const float* emb   = (const float*)d_in[6];
  const float* Wih_f = (const float*)d_in[7];
  const float* Whh_f = (const float*)d_in[8];
  const float* b_f   = (const float*)d_in[9];
  const float* Wih_b = (const float*)d_in[10];
  const float* Whh_b = (const float*)d_in[11];
  const float* b_b   = (const float*)d_in[12];
  const float* Wih_d = (const float*)d_in[13];
  const float* Whh_d = (const float*)d_in[14];
  const float* b_d   = (const float*)d_in[15];
  const float* W_ea  = (const float*)d_in[16];
  const float* W_da  = (const float*)d_in[17];
  const float* W_v   = (const float*)d_in[18];
  const float* b_v   = (const float*)d_in[19];

  char* p = (char*)d_ws;
  auto alloc = [&](size_t bytes) -> char* {
    char* r = p;
    p += (bytes + 255) & ~(size_t)255;
    return r;
  };
  __bf16* xs_bf    = (__bf16*)alloc((size_t)CTE * CB * CE * 2);
  __bf16* decx_bf  = (__bf16*)alloc((size_t)CTD * CB * CE * 2);
  __bf16* Wihf_bf  = (__bf16*)alloc((size_t)G4H * CE * 2);
  __bf16* Whhf_bf  = (__bf16*)alloc((size_t)G4H * CH * 2);
  __bf16* Wihb_bf  = (__bf16*)alloc((size_t)G4H * CE * 2);
  __bf16* Whhb_bf  = (__bf16*)alloc((size_t)G4H * CH * 2);
  __bf16* Wihd_bf  = (__bf16*)alloc((size_t)G4H2 * KIN * 2);
  __bf16* Whhd_bf  = (__bf16*)alloc((size_t)G4H2 * CH2 * 2);
  __bf16* WeaT_bf  = (__bf16*)alloc((size_t)CH2 * CH2 * 2);
  __bf16* WdaT_bf  = (__bf16*)alloc((size_t)CH2 * CH2 * 2);
  __bf16* Wv_bf    = (__bf16*)alloc((size_t)CV * F3 * 2);
  float*  xWf      = (float*)alloc((size_t)CTE * CB * G4H * 4);
  float*  xWb      = (float*)alloc((size_t)CTE * CB * G4H * 4);
  float*  embW     = (float*)alloc((size_t)CTD * CB * G4H2 * 4);
  __bf16* encout_bf= (__bf16*)alloc((size_t)CB * CTE * 2 * CH * 2);
  float*  hcat     = (float*)alloc((size_t)CB * CH2 * 4);
  float*  ccat     = (float*)alloc((size_t)CB * CH2 * 4);
  __bf16* feat     = (__bf16*)alloc((size_t)CTD * CB * F3 * 2);
  float*  hWe      = (float*)alloc((size_t)CB * CH2 * 4);
  float*  hWd      = (float*)alloc((size_t)CB * CH2 * 4);
  float*  sc       = (float*)alloc((size_t)CB * CTE * 4);
  float*  accum    = (float*)alloc((size_t)CB * CTE * 4);
  float*  dsc      = (float*)alloc((size_t)CB * CTD * 4);
  float*  logits   = (float*)alloc((size_t)CTD * CB * CV * 4);
  (void)ws_size; (void)in_sizes; (void)n_in; (void)out_size;

  float* out   = (float*)d_out;
  float* out_h = out + (size_t)CB * CTD * CV;
  float* out_c = out_h + (size_t)CB * CH2;

  auto cvt = [&](const float* src, __bf16* dst, long n) {
    int blocks = (int)((n + 255) / 256);
    if (blocks > 16384) blocks = 16384;
    cvt_bf16_kernel<<<blocks, 256, 0, stream>>>(src, dst, n);
  };
  cvt(Wih_f, Wihf_bf, (long)G4H * CE);
  cvt(Whh_f, Whhf_bf, (long)G4H * CH);
  cvt(Wih_b, Wihb_bf, (long)G4H * CE);
  cvt(Whh_b, Whhb_bf, (long)G4H * CH);
  cvt(Wih_d, Wihd_bf, (long)G4H2 * KIN);
  cvt(Whh_d, Whhd_bf, (long)G4H2 * CH2);
  cvt(W_v,   Wv_bf,   (long)CV * F3);
  cvt_bf16_T_kernel<<<4096, 256, 0, stream>>>(W_ea, WeaT_bf, CH2, CH2);
  cvt_bf16_T_kernel<<<4096, 256, 0, stream>>>(W_da, WdaT_bf, CH2, CH2);

  gather_kernel<<<6400, 256, 0, stream>>>(enc_ids, emb, xs_bf, CTE);
  gather_kernel<<<1024, 256, 0, stream>>>(dec_ids, emb, decx_bf, CTD);

  // bulk input GEMMs (bias folded here); M = 6400 / 1024, both % 64 == 0
  gemm_bf16_kernel<<<dim3(CTE * CB / 64, G4H / 128), 256, 0, stream>>>(
      xs_bf, CE, Wihf_bf, CE, b_f, xWf, G4H, CE);
  gemm_bf16_kernel<<<dim3(CTE * CB / 64, G4H / 128), 256, 0, stream>>>(
      xs_bf, CE, Wihb_bf, CE, b_b, xWb, G4H, CE);
  gemm_bf16_kernel<<<dim3(CTD * CB / 64, G4H2 / 128), 256, 0, stream>>>(
      decx_bf, CE, Wihd_bf /* K-slice [0,256) of ldb=2304 rows */, KIN, b_d, embW, G4H2, CE);

  encoder_lstm_kernel<<<2, 1024, 0, stream>>>(xWf, xWb, Whhf_bf, Whhb_bf, h0, c0,
                                              encout_bf, hcat, ccat);

  decoder_kernel<<<1, 1024, 0, stream>>>(embW, Wihd_bf, Whhd_bf, WeaT_bf, WdaT_bf,
                                         encout_bf, hcat, ccat, feat, hWe, hWd,
                                         sc, accum, dsc, out_h, out_c);

  // dominant GEMM: [1024,3072] x [3072,32000] in bf16 WMMA, 64x128 block tile
  gemm_bf16_kernel<<<dim3(CTD * CB / 64, CV / 128), 256, 0, stream>>>(
      feat, F3, Wv_bf, F3, b_v, logits, CV, F3);

  logsoftmax_kernel<<<CTD * CB, 256, 0, stream>>>(logits, out);
}
